// MVGAT_17463337025762
// MI455X (gfx1250) — matrix operand order repaired and checked by reading, not compiled, not run on hardware
//
#include <hip/hip_runtime.h>
#include <hip/hip_bf16.h>

#define NN 50000
#define EE 800000
#define INDIM 14
#define HID 64
#define NHEAD 2
#define NGRAPH 3
#define NEG_SLOPE 0.2f

typedef __attribute__((ext_vector_type(16))) _Float16 v16h;
typedef __attribute__((ext_vector_type(8)))  float    v8f;

// ---- CDNA5 async global->LDS copy (ASYNCcnt-tracked), per CDNA5 ISA ----
__device__ __forceinline__ void async_g2l_b128(unsigned lds_off, const void* gptr) {
    asm volatile("global_load_async_to_lds_b128 %0, %1, off"
                 :: "v"(lds_off), "v"((unsigned long long)(size_t)gptr)
                 : "memory");
}
__device__ __forceinline__ void wait_async0() {
    asm volatile("s_wait_asynccnt 0x0" ::: "memory");
}

// ---------------- utility kernels ----------------

__global__ void k_fill_f32(float* __restrict__ p, float v, int n) {
    int i = blockIdx.x * blockDim.x + threadIdx.x;
    if (i < n) p[i] = v;
}

__global__ void k_fill_i32(int* __restrict__ p, int v, int n) {
    int i = blockIdx.x * blockDim.x + threadIdx.x;
    if (i < n) p[i] = v;
}

// f32 [rows, cs] -> f16 [rows, cd] zero-padded in K
__global__ void k_cvt_pad(const float* __restrict__ s, _Float16* __restrict__ d,
                          int rows, int cs, int cd) {
    int i = blockIdx.x * blockDim.x + threadIdx.x;
    if (i >= rows * cd) return;
    int r = i / cd, c = i - r * cd;
    d[i] = (c < cs) ? (_Float16)s[(size_t)r * cs + c] : (_Float16)0.0f;
}

// ---------------- WMMA GEMM: D[M,Nout] = X[M,K] * W[Nout,K]^T ----------------
// One workgroup of (Nout/16) waves per 16-row M strip. The whole weight matrix
// (Nout*K f16, <=16KB) is staged into LDS once via async global->LDS b128 copies,
// then each wave computes one 16x16 N-tile with v_wmma_f32_16x16x32_f16 reading
// its B operand from LDS. blockDim.x must be 32*(Nout/16). M%16==0, K%32==0.
__global__ void k_wmma_gemm(const _Float16* __restrict__ X, const _Float16* __restrict__ W,
                            float* __restrict__ D, int K, int Nout) {
    __shared__ __align__(16) _Float16 ldsW[8192];   // 16KB max (Nout*K <= 8192)

    const int tid = threadIdx.x;

    // ---- stage W into LDS (16B per lane per iteration, ASYNCcnt-tracked) ----
    const unsigned ldsBase = (unsigned)(size_t)(&ldsW[0]);   // flat addr low 32b = LDS offset
    const int chunks16 = (Nout * K) >> 3;                    // # of 16-byte chunks
    for (int t = tid; t < chunks16; t += blockDim.x)
        async_g2l_b128(ldsBase + t * 16, (const char*)W + (size_t)t * 16);
    wait_async0();
    __syncthreads();

    // ---- per-wave 16x16 tile ----
    const int wave = tid >> 5;
    const int lane = tid & 31;         // wave32
    const int half = lane >> 4;        // lane group 0/1
    const int idx  = lane & 15;        // row (A) / col (B,D) within tile
    const int m0 = blockIdx.x * 16;
    const int n0 = wave * 16;

    v8f acc = {};
    const _Float16* xrow = X + (size_t)(m0 + idx) * K + half * 8;      // A: global, streamed
    const _Float16* wrow = ldsW + (n0 + idx) * K + half * 8;           // B: from LDS

    for (int k0 = 0; k0 < K; k0 += 32) {
        v16h a, b;
        // A 16x32 f16 layout: lanes0-15 elems0-7 -> K 0..7, elems8-15 -> K 16..23;
        //                     lanes16-31 shifted by +8.
#pragma unroll
        for (int j = 0; j < 8; ++j) {
            a[j]     = xrow[k0 + j];
            a[8 + j] = xrow[k0 + 16 + j];
        }
        // B 32x16 f16 layout: lane holds col n=idx; same K order as A.
#pragma unroll
        for (int j = 0; j < 8; ++j) {
            b[j]     = wrow[k0 + j];
            b[8 + j] = wrow[k0 + 16 + j];
        }

        acc = __builtin_amdgcn_wmma_f32_16x16x32_f16(false, a, false, b,
                                                     (short)0, acc, false, false);
    }
    // D 16x16 f32: VGPR r -> row = half*8 + r, col = idx
#pragma unroll
    for (int r = 0; r < 8; ++r)
        D[(size_t)(m0 + half * 8 + r) * Nout + n0 + idx] = acc[r];
}

// ---------------- attention logits: el/er[n,h] = dot(ft[n,h,:], al/ar[h,:]) ----------------
__global__ void k_logits(const float* __restrict__ ft, const float* __restrict__ al,
                         const float* __restrict__ ar, float* __restrict__ el,
                         float* __restrict__ er, int Dh, int total) {
    int i = blockIdx.x * blockDim.x + threadIdx.x;
    if (i >= total) return;               // total = NN * NHEAD
    int n = i >> 1, h = i & 1;
    const float* f = ft + (size_t)n * (2 * Dh) + h * Dh;
    const float* a = al + h * Dh;
    const float* b = ar + h * Dh;
    float sl = 0.f, sr = 0.f;
    for (int j = 0; j < Dh; ++j) { float v = f[j]; sl += v * a[j]; sr += v * b[j]; }
    el[i] = sl; er[i] = sr;
}

// ---------------- edge pass 1: e = lrelu(el[src]+er[dst]); segment max over dst ----------------
__global__ void k_edge1(const int* __restrict__ src, const int* __restrict__ dst,
                        const float* __restrict__ el, const float* __restrict__ er,
                        float* __restrict__ ebuf, int* __restrict__ mkey, int total) {
    int i = blockIdx.x * blockDim.x + threadIdx.x;
    if (i >= total) return;               // total = EE * NHEAD
    int e = i >> 1, h = i & 1;
    int s = src[e], d = dst[e];
    float v = el[s * 2 + h] + er[d * 2 + h];
    v = v > 0.f ? v : NEG_SLOPE * v;
    ebuf[i] = v;
    int b = __float_as_int(v);
    int key = (b >= 0) ? b : (b ^ 0x7fffffff);   // order-preserving int mapping
    atomicMax(&mkey[d * 2 + h], key);
}

// ---------------- edge pass 2: ex = exp(e - m[dst]); segment sum over dst ----------------
__global__ void k_edge2(const int* __restrict__ dst, float* __restrict__ ebuf,
                        const int* __restrict__ mkey, float* __restrict__ ssum, int total) {
    int i = blockIdx.x * blockDim.x + threadIdx.x;
    if (i >= total) return;
    int e = i >> 1, h = i & 1;
    int d = dst[e];
    int k = mkey[d * 2 + h];
    k = (k >= 0) ? k : (k ^ 0x7fffffff);
    float ex = expf(ebuf[i] - __int_as_float(k));
    ebuf[i] = ex;
    atomicAdd(&ssum[d * 2 + h], ex);
}

// ---------------- edge pass 3: acc[dst,h,:] += ft[src,h,:] * (ex / s[dst,h]) ----------------
// One thread per (edge, head, 4-channel chunk): 16x the threads of the naive
// version, same atomic count, much better latency hiding on the L2 atomic pipes.
__global__ void k_edge3(const int* __restrict__ src, const int* __restrict__ dst,
                        const float* __restrict__ ebuf, const float* __restrict__ ssum,
                        const float* __restrict__ ft, float* __restrict__ acc,
                        int Dh, int total) {
    int i = blockIdx.x * blockDim.x + threadIdx.x;
    if (i >= total) return;               // total = EE * NHEAD * (Dh/4)
    int chunks = Dh >> 2;
    int c = i % chunks;
    int rest = i / chunks;                // rest = e*2 + h
    int h = rest & 1, e = rest >> 1;
    int s = src[e], d = dst[e];
    float alpha = ebuf[rest] / ssum[d * 2 + h];
    const float* fs = ft + (size_t)s * (2 * Dh) + h * Dh + c * 4;
    float* ad = acc + (size_t)d * (2 * Dh) + h * Dh + c * 4;
#pragma unroll
    for (int j = 0; j < 4; ++j) atomicAdd(&ad[j], fs[j] * alpha);
}

// ---------------- finalize: out = relu(acc + res + bias) ----------------
__global__ void k_final(const float* __restrict__ acc, const float* __restrict__ res,
                        const float* __restrict__ bias, float* __restrict__ out,
                        int HD, int n) {
    int i = blockIdx.x * blockDim.x + threadIdx.x;
    if (i >= n) return;
    int c = i % HD;
    float v = acc[i] + res[i] + bias[c];
    out[i] = v > 0.f ? v : 0.f;
}

// ---------------- host launcher ----------------

static inline int cdiv(long long a, long long b) { return (int)((a + b - 1) / b); }

extern "C" void kernel_launch(void* const* d_in, const int* in_sizes, int n_in,
                              void* d_out, int out_size, void* d_ws, size_t ws_size,
                              hipStream_t stream) {
    const float* feat = (const float*)d_in[0];
    const int* srcs[3] = { (const int*)d_in[1], (const int*)d_in[3], (const int*)d_in[5] };
    const int* dsts[3] = { (const int*)d_in[2], (const int*)d_in[4], (const int*)d_in[6] };
    const float* W0  = (const float*)d_in[7];
    const float* al0 = (const float*)d_in[8];
    const float* ar0 = (const float*)d_in[9];
    const float* b0  = (const float*)d_in[10];
    const float* rW0 = (const float*)d_in[11];
    const float* W1  = (const float*)d_in[12];
    const float* al1 = (const float*)d_in[13];
    const float* ar1 = (const float*)d_in[14];
    const float* b1  = (const float*)d_in[15];
    const float* rW1 = (const float*)d_in[16];
    float* out = (float*)d_out;

    char* ws = (char*)d_ws;
    size_t off = 0;
    auto take = [&](size_t bytes) -> void* {
        void* p = (void*)(ws + off);
        off = (off + bytes + 255) & ~(size_t)255;
        return p;
    };
    _Float16* featH = (_Float16*)take((size_t)NN * 32 * 2);    // feat padded 14->32
    _Float16* zH    = (_Float16*)take((size_t)NN * 128 * 2);   // layer-1 input
    _Float16* W0h   = (_Float16*)take((size_t)128 * 32 * 2);
    _Float16* rW0h  = (_Float16*)take((size_t)128 * 32 * 2);
    _Float16* W1h   = (_Float16*)take((size_t)64 * 128 * 2);
    _Float16* rW1h  = (_Float16*)take((size_t)64 * 128 * 2);
    float* ftb  = (float*)take((size_t)NN * 128 * 4);          // ft, later reused for z
    float* resb = (float*)take((size_t)NN * 128 * 4);          // residual projection
    float* accb = (float*)take((size_t)NN * 128 * 4);          // message accumulator
    float* elb  = (float*)take((size_t)NN * 2 * 4);
    float* erb  = (float*)take((size_t)NN * 2 * 4);
    float* ssum = (float*)take((size_t)NN * 2 * 4);
    int*   mkey = (int*)take((size_t)NN * 2 * 4);
    float* ebuf = (float*)take((size_t)EE * 2 * 4);

    const int TB = 256;
    const int EH = EE * 2;

    // feat -> f16 once (shared by all views)
    k_cvt_pad<<<cdiv((long long)NN * 32, TB), TB, 0, stream>>>(feat, featH, NN, INDIM, 32);

    for (int g = 0; g < NGRAPH; ++g) {
        const int* src = srcs[g];
        const int* dst = dsts[g];

        // ================= Layer 0 (in 14 -> 2 heads x 64) =================
        k_cvt_pad<<<cdiv(128 * 32, TB), TB, 0, stream>>>(W0  + (size_t)g * 128 * INDIM, W0h,  128, INDIM, 32);
        k_cvt_pad<<<cdiv(128 * 32, TB), TB, 0, stream>>>(rW0 + (size_t)g * 128 * INDIM, rW0h, 128, INDIM, 32);

        // 8 waves/block (Nout=128): block = 256
        k_wmma_gemm<<<NN / 16, 256, 0, stream>>>(featH, W0h,  ftb,  32, 128);
        k_wmma_gemm<<<NN / 16, 256, 0, stream>>>(featH, rW0h, resb, 32, 128);

        k_logits<<<cdiv(NN * 2, TB), TB, 0, stream>>>(ftb, al0 + (size_t)g * 2 * HID,
                                                      ar0 + (size_t)g * 2 * HID, elb, erb, HID, NN * 2);
        k_fill_i32<<<cdiv(NN * 2, TB), TB, 0, stream>>>(mkey, (int)0x80000000, NN * 2);
        k_fill_f32<<<cdiv(NN * 2, TB), TB, 0, stream>>>(ssum, 0.f, NN * 2);
        k_fill_f32<<<cdiv((long long)NN * 128, TB), TB, 0, stream>>>(accb, 0.f, NN * 128);

        k_edge1<<<cdiv(EH, TB), TB, 0, stream>>>(src, dst, elb, erb, ebuf, mkey, EH);
        k_edge2<<<cdiv(EH, TB), TB, 0, stream>>>(dst, ebuf, mkey, ssum, EH);
        k_edge3<<<cdiv((long long)EH * 16, TB), TB, 0, stream>>>(src, dst, ebuf, ssum, ftb, accb,
                                                                 HID, EH * 16);

        // z0 = relu(acc + res + b0); write into ftb (ft no longer needed)
        k_final<<<cdiv((long long)NN * 128, TB), TB, 0, stream>>>(accb, resb, b0 + (size_t)g * 128,
                                                                 ftb, 128, NN * 128);
        k_cvt_pad<<<cdiv((long long)NN * 128, TB), TB, 0, stream>>>(ftb, zH, NN, 128, 128);

        // ================= Layer 1 (in 128 -> 2 heads x 32) =================
        k_cvt_pad<<<cdiv(64 * 128, TB), TB, 0, stream>>>(W1  + (size_t)g * 64 * 128, W1h,  64, 128, 128);
        k_cvt_pad<<<cdiv(64 * 128, TB), TB, 0, stream>>>(rW1 + (size_t)g * 64 * 128, rW1h, 64, 128, 128);

        // 4 waves/block (Nout=64): block = 128
        k_wmma_gemm<<<NN / 16, 128, 0, stream>>>(zH, W1h,  ftb,  128, 64);
        k_wmma_gemm<<<NN / 16, 128, 0, stream>>>(zH, rW1h, resb, 128, 64);

        k_logits<<<cdiv(NN * 2, TB), TB, 0, stream>>>(ftb, al1 + (size_t)g * 64,
                                                      ar1 + (size_t)g * 64, elb, erb, 32, NN * 2);
        k_fill_i32<<<cdiv(NN * 2, TB), TB, 0, stream>>>(mkey, (int)0x80000000, NN * 2);
        k_fill_f32<<<cdiv(NN * 2, TB), TB, 0, stream>>>(ssum, 0.f, NN * 2);
        k_fill_f32<<<cdiv((long long)NN * 64, TB), TB, 0, stream>>>(accb, 0.f, NN * 64);

        k_edge1<<<cdiv(EH, TB), TB, 0, stream>>>(src, dst, elb, erb, ebuf, mkey, EH);
        k_edge2<<<cdiv(EH, TB), TB, 0, stream>>>(dst, ebuf, mkey, ssum, EH);
        k_edge3<<<cdiv((long long)EH * 8, TB), TB, 0, stream>>>(src, dst, ebuf, ssum, ftb, accb,
                                                                32, EH * 8);

        k_final<<<cdiv((long long)NN * 64, TB), TB, 0, stream>>>(accb, resb, b1 + (size_t)g * 64,
                                                                 out + (size_t)g * NN * 64, 64, NN * 64);
    }
}
